// Flashback_86363202387996
// MI455X (gfx1250) — matrix operand order repaired and checked by reading, not compiled, not run on hardware
//
#include <hip/hip_runtime.h>
#include <hip/hip_bf16.h>
#include <math.h>

// ---------------- problem constants ----------------
#define SEQ    20
#define USERS  256
#define NLOC   20000
#define HID    250
#define DIMX   256
#define NHEADS 8
#define DH     32
#define FFND   1024
#define MTOT   (USERS*SEQ)   // 5120 rows for all sequence GEMMs
#define K2H    500           // 2*HID
#define K2HP   512           // padded K for final GEMM

typedef __attribute__((ext_vector_type(16))) _Float16 v16h;
typedef __attribute__((ext_vector_type(8)))  float    v8f;
typedef _Float16 hf;

static __device__ __forceinline__ float gelu_tanh(float x) {
  const float c = 0.7978845608028654f; // sqrt(2/pi)
  float x3 = x * x * x;
  return 0.5f * x * (1.0f + tanhf(c * (x + 0.044715f * x3)));
}

template <int BS>
static __device__ __forceinline__ float block_sum(float v, float* red) {
  int t = threadIdx.x;
  red[t] = v; __syncthreads();
  #pragma unroll
  for (int s = BS / 2; s > 0; s >>= 1) {
    if (t < s) red[t] += red[t + s];
    __syncthreads();
  }
  float r = red[0]; __syncthreads();
  return r;
}

// ---------------------------------------------------------------
// 1) Build fused-linear input: [emb[x] | time2vec] as f16, m = u*SEQ+s
// ---------------------------------------------------------------
__global__ void build_fuse_in(const int* __restrict__ x, const int* __restrict__ tslot,
                              const float* __restrict__ emb,
                              const float* __restrict__ t2v_w, const float* __restrict__ t2v_b,
                              const float* __restrict__ t2v_w0, const float* __restrict__ t2v_b0,
                              hf* __restrict__ A) {
  int m = blockIdx.x;            // 0..5119
  int j = threadIdx.x;           // 0..255
  int u = m / SEQ, s = m % SEQ;
  float v;
  if (j < HID) {
    int loc = x[s * USERS + u];
    v = emb[(size_t)loc * HID + j];
  } else {
    float tau = (float)tslot[s * USERS + u] / 168.0f;
    if (j < DIMX - 1) {
      int q = j - HID;           // 0..4
      v = sinf(tau * t2v_w[q] + t2v_b[q]);
    } else {
      v = tau * t2v_w0[0] + t2v_b0[0];
    }
  }
  A[(size_t)m * DIMX + j] = (hf)v;
}

// ---------------------------------------------------------------
// 2) f32 -> f16 with K padding (weights + activations staging)
// ---------------------------------------------------------------
__global__ void cvt_f16_pad(const float* __restrict__ src, hf* __restrict__ dst,
                            int rows, int K, int Kpad) {
  size_t idx = (size_t)blockIdx.x * blockDim.x + threadIdx.x;
  size_t total = (size_t)rows * Kpad;
  if (idx >= total) return;
  int c = (int)(idx % Kpad);
  size_t r = idx / Kpad;
  dst[idx] = (c < K) ? (hf)src[r * K + c] : (hf)0.0f;
}

// ---------------------------------------------------------------
// 3) Generic WMMA GEMM: C[M,N] = act(A[M,Kp] @ B[N,Kp]^T + bias) + residual
//    8 waves/block, wave-per-16x16-tile, v_wmma_f32_16x16x32_f16
// ---------------------------------------------------------------
__global__ __launch_bounds__(256) void wmma_gemm(const hf* __restrict__ A, const hf* __restrict__ B,
                                                 const float* __restrict__ bias,
                                                 const float* __restrict__ residual,
                                                 float* __restrict__ C,
                                                 int M, int N, int Kpad, int act) {
  int wave = threadIdx.x >> 5;
  int lane = threadIdx.x & 31;
  int n0 = (blockIdx.x * 8 + wave) * 16;
  int m0 = blockIdx.y * 16;
  if (n0 >= N) return;                       // wave-uniform exit: EXEC stays full
  int l15 = lane & 15;
  int hw  = lane >> 4;                       // half-wave select
  const hf* arow = A + (size_t)(m0 + l15) * Kpad;
  int ncol = n0 + l15;
  int ncl  = ncol < N ? ncol : N - 1;        // clamp loads, guard stores
  const hf* brow = B + (size_t)ncl * Kpad;

  v8f acc = {};
  for (int k0 = 0; k0 < Kpad; k0 += 32) {
    v16h a, b;
    int ka = k0 + hw * 8;
    #pragma unroll
    for (int j = 0; j < 8; ++j) a[j] = arow[ka + j];            // K = ka+0..7
    #pragma unroll
    for (int j = 0; j < 8; ++j) a[8 + j] = arow[ka + 16 + j];   // K = ka+16..23
    int kb = k0 + hw * 8;
    #pragma unroll
    for (int j = 0; j < 8; ++j) b[j] = brow[kb + j];            // K = kb+0..7
    #pragma unroll
    for (int j = 0; j < 8; ++j) b[8 + j] = brow[kb + 16 + j];   // K = kb+16..23
    acc = __builtin_amdgcn_wmma_f32_16x16x32_f16(false, a, false, b,
                                                 (short)0, acc, false, false);
  }

  if (ncol < N) {
    float bs = bias ? bias[ncl] : 0.0f;
    #pragma unroll
    for (int r = 0; r < 8; ++r) {
      int m = m0 + r + hw * 8;
      float v = acc[r] + bs;
      if (act == 1)      v = v > 0.0f ? v : 0.2f * v;   // LeakyReLU(0.2)
      else if (act == 2) v = gelu_tanh(v);
      if (residual) v += residual[(size_t)m * N + ncol];
      C[(size_t)m * N + ncol] = v;
    }
  }
}

// ---------------------------------------------------------------
// 4) LayerNorm over last dim (256), emits f16 staging directly
// ---------------------------------------------------------------
__global__ void ln_to_f16(const float* __restrict__ X, const float* __restrict__ g,
                          const float* __restrict__ b, hf* __restrict__ Y) {
  __shared__ float red[DIMX];
  int m = blockIdx.x, t = threadIdx.x;
  float v = X[(size_t)m * DIMX + t];
  float mean = block_sum<DIMX>(v, red) * (1.0f / DIMX);
  float d = v - mean;
  float var = block_sum<DIMX>(d * d, red) * (1.0f / DIMX);
  float y = d * rsqrtf(var + 1e-5f) * g[t] + b[t];
  Y[(size_t)m * DIMX + t] = (hf)y;
}

// ---------------------------------------------------------------
// 5) Per-(user,head) attention, S=20, dh=32 (tiny)
// ---------------------------------------------------------------
__global__ void attn_kernel(const float* __restrict__ Q, const float* __restrict__ K,
                            const float* __restrict__ V, float* __restrict__ CTX) {
  int u = blockIdx.x, h = blockIdx.y;
  int i = threadIdx.x;
  if (i >= SEQ) return;
  const float scale = 0.17677669529663687f; // 1/sqrt(32)
  const float* q = Q + ((size_t)u * SEQ + i) * DIMX + h * DH;
  float sc[SEQ], mx = -3.0e38f;
  for (int j = 0; j < SEQ; ++j) {
    const float* k = K + ((size_t)u * SEQ + j) * DIMX + h * DH;
    float d = 0.0f;
    #pragma unroll
    for (int e = 0; e < DH; ++e) d += q[e] * k[e];
    sc[j] = d * scale;
    mx = fmaxf(mx, sc[j]);
  }
  float den = 0.0f;
  for (int j = 0; j < SEQ; ++j) { sc[j] = expf(sc[j] - mx); den += sc[j]; }
  float inv = 1.0f / den;
  float* o = CTX + ((size_t)u * SEQ + i) * DIMX + h * DH;
  #pragma unroll
  for (int e = 0; e < DH; ++e) {
    float a = 0.0f;
    for (int j = 0; j < SEQ; ++j)
      a += sc[j] * V[((size_t)u * SEQ + j) * DIMX + h * DH + e];
    o[e] = a * inv;
  }
}

// ---------------------------------------------------------------
// 6) out_time decoder (DIM -> 1), layout (S,U)
// ---------------------------------------------------------------
__global__ void dec_time(const float* __restrict__ ENC, const float* __restrict__ W,
                         const float* __restrict__ b, float* __restrict__ OT) {
  int m = blockIdx.x * blockDim.x + threadIdx.x;
  if (m >= MTOT) return;
  int u = m / SEQ, s = m % SEQ;
  const float* e = ENC + (size_t)m * DIMX;
  float d = 0.0f;
  for (int j = 0; j < DIMX; ++j) d += e[j] * W[j];
  OT[s * USERS + u] = d + b[0];
}

// ---------------------------------------------------------------
// 7) Haversine decay weighting + concat user-emb -> out_pu (f16, Kpad=512)
//    also computes 1/||row|| for the cosine epilogue. m' = s*USERS+u
// ---------------------------------------------------------------
__global__ void build_outpu(const float* __restrict__ OUT /* (u*SEQ+j)*HID */,
                            const float* __restrict__ sc /* (S,U,2) */,
                            const float* __restrict__ user_emb,
                            const int* __restrict__ active_user,
                            hf* __restrict__ Apu, float* __restrict__ rnA) {
  __shared__ float red[K2HP];
  int i = blockIdx.x;   // seq index
  int u = blockIdx.y;   // user
  int t = threadIdx.x;  // 0..511
  const float d2r = 0.017453292519943295f;
  float lat_i = sc[((size_t)i * USERS + u) * 2 + 0] * d2r;
  float lon_i = sc[((size_t)i * USERS + u) * 2 + 1] * d2r;
  float val = 0.0f;
  if (t < HID) {
    float acc = 0.0f, wsum = 0.0f;
    for (int j = 0; j <= i; ++j) {
      float lat_j = sc[((size_t)j * USERS + u) * 2 + 0] * d2r;
      float lon_j = sc[((size_t)j * USERS + u) * 2 + 1] * d2r;
      float sdlat = sinf((lat_j - lat_i) * 0.5f);
      float sdlon = sinf((lon_j - lon_i) * 0.5f);
      float a = sdlat * sdlat + cosf(lat_i) * cosf(lat_j) * sdlon * sdlon;
      a = fminf(fmaxf(a, 1e-12f), 1.0f);
      float dist = 2.0f * 6371.0f * asinf(sqrtf(a));
      float w = expf(-0.1f * dist) + 1e-10f;
      acc += w * OUT[((size_t)u * SEQ + j) * HID + t];
      wsum += w;
    }
    val = acc / wsum;
  } else if (t < K2H) {
    val = user_emb[(size_t)active_user[u] * HID + (t - HID)];
  }
  int m = i * USERS + u;
  Apu[(size_t)m * K2HP + t] = (hf)val;            // t >= 500 pads zero
  float ss = block_sum<K2HP>(val * val, red);
  if (t == 0) rnA[m] = 1.0f / fmaxf(sqrtf(ss), 1e-12f);
}

// ---------------------------------------------------------------
// 8) 1/||fc_W row|| (K=500)
// ---------------------------------------------------------------
__global__ void rownorm_w(const float* __restrict__ W, float* __restrict__ rnB) {
  __shared__ float red[256];
  int n = blockIdx.x, t = threadIdx.x;
  const float* r = W + (size_t)n * K2H;
  float v = 0.0f;
  if (t < K2H)       { float a = r[t];       v += a * a; }
  if (t + 256 < K2H) { float a = r[t + 256]; v += a * a; }
  float ss = block_sum<256>(v, red);
  if (t == 0) rnB[n] = 1.0f / fmaxf(sqrtf(ss), 1e-12f);
}

// ---------------------------------------------------------------
// 9) Final FC via WMMA: one GEMM feeds y_linear AND cosine
// ---------------------------------------------------------------
__global__ __launch_bounds__(256) void fc_wmma(const hf* __restrict__ A, const hf* __restrict__ B,
                                               const float* __restrict__ bias,
                                               const float* __restrict__ rnA,
                                               const float* __restrict__ rnB,
                                               float* __restrict__ YL, float* __restrict__ CS) {
  int wave = threadIdx.x >> 5;
  int lane = threadIdx.x & 31;
  int n0 = (blockIdx.x * 8 + wave) * 16;
  int m0 = blockIdx.y * 16;
  if (n0 >= NLOC) return;
  int l15 = lane & 15;
  int hw  = lane >> 4;
  const hf* arow = A + (size_t)(m0 + l15) * K2HP;
  int ncol = n0 + l15;
  int ncl  = ncol < NLOC ? ncol : NLOC - 1;
  const hf* brow = B + (size_t)ncl * K2HP;

  v8f acc = {};
  #pragma unroll
  for (int k0 = 0; k0 < K2HP; k0 += 32) {
    if (k0 + 64 < K2HP) __builtin_prefetch(brow + k0 + 64, 0, 1);  // global_prefetch_b8
    v16h a, b;
    int ka = k0 + hw * 8;
    #pragma unroll
    for (int j = 0; j < 8; ++j) a[j] = arow[ka + j];
    #pragma unroll
    for (int j = 0; j < 8; ++j) a[8 + j] = arow[ka + 16 + j];
    int kb = k0 + hw * 8;
    #pragma unroll
    for (int j = 0; j < 8; ++j) b[j] = brow[kb + j];
    #pragma unroll
    for (int j = 0; j < 8; ++j) b[8 + j] = brow[kb + 16 + j];
    acc = __builtin_amdgcn_wmma_f32_16x16x32_f16(false, a, false, b,
                                                 (short)0, acc, false, false);
  }

  if (ncol < NLOC) {
    float bs = bias[ncl];
    float rb = rnB[ncl];
    #pragma unroll
    for (int r = 0; r < 8; ++r) {
      int m = m0 + r + hw * 8;
      float z = acc[r];
      size_t idx = (size_t)m * NLOC + ncol;
      YL[idx] = z + bs;
      CS[idx] = z * rnA[m] * rb;
    }
  }
}

// ================================================================
// host side
// ================================================================
static inline int idiv_up(long long a, long long b) { return (int)((a + b - 1) / b); }

extern "C" void kernel_launch(void* const* d_in, const int* in_sizes, int n_in,
                              void* d_out, int out_size, void* d_ws, size_t ws_size,
                              hipStream_t stream) {
  // inputs
  const int*   x         = (const int*)  d_in[0];
  const int*   t_slot    = (const int*)  d_in[1];
  const float* s_coord   = (const float*)d_in[2];
  const float* h_in      = (const float*)d_in[3];
  const int*   act_user  = (const int*)  d_in[4];
  const float* emb       = (const float*)d_in[5];
  const float* user_emb  = (const float*)d_in[6];
  const float* t2v_w     = (const float*)d_in[7];
  const float* t2v_b     = (const float*)d_in[8];
  const float* t2v_w0    = (const float*)d_in[9];
  const float* t2v_b0    = (const float*)d_in[10];
  const float* fuse_W    = (const float*)d_in[11];
  const float* fuse_b    = (const float*)d_in[12];
  const float* ln1_g     = (const float*)d_in[13];
  const float* ln1_b     = (const float*)d_in[14];
  const float* Wq        = (const float*)d_in[15];
  const float* bq        = (const float*)d_in[16];
  const float* Wk        = (const float*)d_in[17];
  const float* bk        = (const float*)d_in[18];
  const float* Wv        = (const float*)d_in[19];
  const float* bv        = (const float*)d_in[20];
  const float* Wo        = (const float*)d_in[21];
  const float* bo        = (const float*)d_in[22];
  const float* ln2_g     = (const float*)d_in[23];
  const float* ln2_b     = (const float*)d_in[24];
  const float* ffn_W1    = (const float*)d_in[25];
  const float* ffn_b1    = (const float*)d_in[26];
  const float* ffn_W2    = (const float*)d_in[27];
  const float* ffn_b2    = (const float*)d_in[28];
  const float* dec_W     = (const float*)d_in[29];
  const float* dec_b     = (const float*)d_in[30];
  const float* dec_t_W   = (const float*)d_in[31];
  const float* dec_t_b   = (const float*)d_in[32];
  const float* fc_W      = (const float*)d_in[33];
  const float* fc_b      = (const float*)d_in[34];

  // outputs (flat, return order: y_linear, h, cosine, out_time)
  float* out_f = (float*)d_out;
  const size_t YL_OFF = 0;
  const size_t YL_SZ  = (size_t)SEQ * USERS * NLOC;     // 102,400,000
  const size_t H_OFF  = YL_OFF + YL_SZ;
  const size_t H_SZ   = (size_t)USERS * HID;            // 64,000
  const size_t CS_OFF = H_OFF + H_SZ;
  const size_t OT_OFF = CS_OFF + YL_SZ;
  float* d_yl = out_f + YL_OFF;
  float* d_cs = out_f + CS_OFF;
  float* d_ot = out_f + OT_OFF;

  // workspace bump allocator (256B aligned)
  char* w = (char*)d_ws;
  size_t off = 0;
  auto alloc = [&](size_t bytes) -> void* {
    void* p = w + off;
    off += (bytes + 255) & ~(size_t)255;
    return p;
  };
  // f32 activations
  float* fused = (float*)alloc((size_t)MTOT * DIMX * 4);
  float* qf    = (float*)alloc((size_t)MTOT * DIMX * 4);
  float* kf    = (float*)alloc((size_t)MTOT * DIMX * 4);
  float* vf    = (float*)alloc((size_t)MTOT * DIMX * 4);
  float* ctx   = (float*)alloc((size_t)MTOT * DIMX * 4);
  float* xx    = (float*)alloc((size_t)MTOT * DIMX * 4);
  float* ff1   = (float*)alloc((size_t)MTOT * FFND * 4);
  float* enc   = (float*)alloc((size_t)MTOT * DIMX * 4);
  float* outb  = (float*)alloc((size_t)MTOT * HID  * 4);
  float* rnA   = (float*)alloc((size_t)MTOT * 4);
  float* rnB   = (float*)alloc((size_t)NLOC * 4);
  // f16 stagings
  hf* Afuse = (hf*)alloc((size_t)MTOT * DIMX * 2);
  hf* h1h   = (hf*)alloc((size_t)MTOT * DIMX * 2);
  hf* ctxh  = (hf*)alloc((size_t)MTOT * DIMX * 2);
  hf* h2h   = (hf*)alloc((size_t)MTOT * DIMX * 2);
  hf* ff1h  = (hf*)alloc((size_t)MTOT * FFND * 2);
  hf* ench  = (hf*)alloc((size_t)MTOT * DIMX * 2);
  hf* Apu   = (hf*)alloc((size_t)MTOT * K2HP * 2);
  // f16 weights
  hf* Wfh  = (hf*)alloc((size_t)DIMX * DIMX * 2);
  hf* Wqh  = (hf*)alloc((size_t)DIMX * DIMX * 2);
  hf* Wkh  = (hf*)alloc((size_t)DIMX * DIMX * 2);
  hf* Wvh  = (hf*)alloc((size_t)DIMX * DIMX * 2);
  hf* Woh  = (hf*)alloc((size_t)DIMX * DIMX * 2);
  hf* W1h  = (hf*)alloc((size_t)FFND * DIMX * 2);
  hf* W2h  = (hf*)alloc((size_t)DIMX * FFND * 2);
  hf* Wdh  = (hf*)alloc((size_t)HID  * DIMX * 2);
  hf* Wfch = (hf*)alloc((size_t)NLOC * K2HP * 2);
  (void)ws_size; (void)in_sizes; (void)n_in; (void)out_size;

  // ---- weight conversions to f16 ----
  auto cvt = [&](const float* src, hf* dst, int rows, int K, int Kpad) {
    long long tot = (long long)rows * Kpad;
    cvt_f16_pad<<<idiv_up(tot, 256), 256, 0, stream>>>(src, dst, rows, K, Kpad);
  };
  cvt(fuse_W, Wfh, DIMX, DIMX, DIMX);
  cvt(Wq, Wqh, DIMX, DIMX, DIMX);
  cvt(Wk, Wkh, DIMX, DIMX, DIMX);
  cvt(Wv, Wvh, DIMX, DIMX, DIMX);
  cvt(Wo, Woh, DIMX, DIMX, DIMX);
  cvt(ffn_W1, W1h, FFND, DIMX, DIMX);
  cvt(ffn_W2, W2h, DIMX, FFND, FFND);
  cvt(dec_W, Wdh, HID, DIMX, DIMX);
  cvt(fc_W, Wfch, NLOC, K2H, K2HP);

  // ---- embeddings + time2vec ----
  build_fuse_in<<<MTOT, DIMX, 0, stream>>>(x, t_slot, emb, t2v_w, t2v_b, t2v_w0, t2v_b0, Afuse);

  auto gemm = [&](const hf* A, const hf* B, const float* bias, const float* res,
                  float* C, int M, int N, int Kpad, int actk) {
    dim3 grid(idiv_up(idiv_up(N, 16), 8), M / 16);
    wmma_gemm<<<grid, 256, 0, stream>>>(A, B, bias, res, C, M, N, Kpad, actk);
  };

  // fused = LeakyReLU(A @ fuse_W^T + b)
  gemm(Afuse, Wfh, fuse_b, nullptr, fused, MTOT, DIMX, DIMX, 1);
  // h1 = LN(fused) -> f16
  ln_to_f16<<<MTOT, DIMX, 0, stream>>>(fused, ln1_g, ln1_b, h1h);
  // q,k,v
  gemm(h1h, Wqh, bq, nullptr, qf, MTOT, DIMX, DIMX, 0);
  gemm(h1h, Wkh, bk, nullptr, kf, MTOT, DIMX, DIMX, 0);
  gemm(h1h, Wvh, bv, nullptr, vf, MTOT, DIMX, DIMX, 0);
  // attention
  attn_kernel<<<dim3(USERS, NHEADS), 32, 0, stream>>>(qf, kf, vf, ctx);
  cvt(ctx, ctxh, MTOT, DIMX, DIMX);
  // xx = fused + ctx @ Wo^T + bo
  gemm(ctxh, Woh, bo, fused, xx, MTOT, DIMX, DIMX, 0);
  // h2 = LN(xx)
  ln_to_f16<<<MTOT, DIMX, 0, stream>>>(xx, ln2_g, ln2_b, h2h);
  // FFN
  gemm(h2h, W1h, ffn_b1, nullptr, ff1, MTOT, FFND, DIMX, 2);   // gelu
  cvt(ff1, ff1h, MTOT, FFND, FFND);
  gemm(ff1h, W2h, ffn_b2, xx, enc, MTOT, DIMX, FFND, 0);       // + residual
  cvt(enc, ench, MTOT, DIMX, DIMX);
  // decoders
  gemm(ench, Wdh, dec_b, nullptr, outb, MTOT, HID, DIMX, 0);
  dec_time<<<idiv_up(MTOT, 256), 256, 0, stream>>>(enc, dec_t_W, dec_t_b, d_ot);
  // spatial-decay weighting + concat user emb + row norms
  build_outpu<<<dim3(SEQ, USERS), K2HP, 0, stream>>>(outb, s_coord, user_emb, act_user, Apu, rnA);
  rownorm_w<<<NLOC, 256, 0, stream>>>(fc_W, rnB);
  // final FC: one WMMA GEMM -> y_linear + cosine
  {
    dim3 grid(idiv_up(idiv_up(NLOC, 16), 8), MTOT / 16);
    fc_wmma<<<grid, 256, 0, stream>>>(Apu, Wfch, fc_b, rnA, rnB, d_yl, d_cs);
  }
  // h passthrough
  hipMemcpyAsync(out_f + H_OFF, h_in, H_SZ * sizeof(float),
                 hipMemcpyDeviceToDevice, stream);
}
